// GraphQNetwork_76974403879152
// MI455X (gfx1250) — matrix operand order repaired and checked by reading, not compiled, not run on hardware
//
#include <hip/hip_runtime.h>
#include <hip/hip_bf16.h>
#include <math.h>

#define N_NODES 50000
#define N_EDGES 800000
#define F 128

typedef __attribute__((ext_vector_type(16))) __bf16 v16bf;
typedef __attribute__((ext_vector_type(8)))  float  v8f;

#if defined(__has_builtin)
#  if __has_builtin(__builtin_amdgcn_global_load_async_to_lds_b128)
#    define HAVE_ASYNC_COPY 1
#  endif
#  if __has_builtin(__builtin_amdgcn_s_wait_asynccnt)
#    define HAVE_WAIT_ASYNC_BUILTIN 1
#  endif
#endif

#if defined(HAVE_ASYNC_COPY)
typedef int v4i_ __attribute__((vector_size(16)));
typedef __attribute__((address_space(1))) v4i_* g4p;   // global int4*
typedef __attribute__((address_space(3))) v4i_* l4p;   // LDS int4*

__device__ inline void wait_asynccnt0() {
#if defined(HAVE_WAIT_ASYNC_BUILTIN)
  __builtin_amdgcn_s_wait_asynccnt(0);
#else
  asm volatile("s_wait_asynccnt 0x0" ::: "memory");
#endif
}
#endif

__device__ inline unsigned int pack_bf16(float a, float b) {
  union { __bf16 h; unsigned short s; } ua, ub;
  ua.h = (__bf16)a; ub.h = (__bf16)b;
  return (unsigned int)ua.s | ((unsigned int)ub.s << 16);
}

// One-shot: W[k][n] fp32 (128x128) -> Wt packed bf16 pairs, Wt[n*64+kp] = {W[2kp][n], W[2kp+1][n]}
__global__ __launch_bounds__(256) void prep_weight_kernel(
    const float* __restrict__ W, unsigned int* __restrict__ Wt)
{
  int i = (int)blockIdx.x * 256 + (int)threadIdx.x;   // 0..8191
  if (i < 128 * 64) {
    int n = i >> 6, kp = i & 63;
    Wt[i] = pack_bf16(W[(size_t)(2 * kp) * F + n], W[(size_t)(2 * kp + 1) * F + n]);
  }
}

// out[n x 128] = relu( A0 @ W0 (+ A1 @ W1) + bias ), bf16 WMMA, f32 accumulate.
// Wt0/Wt1 are pre-transposed packed-bf16 weights (prep_weight_kernel layout).
// Block: 256 threads = 8 waves; 64 rows x 128 cols per block; wave w owns col tile w.
__global__ __launch_bounds__(256) void sage_gemm_kernel(
    const float* __restrict__ A0, const unsigned int* __restrict__ Wt0,
    const float* __restrict__ A1, const unsigned int* __restrict__ Wt1,
    const float* __restrict__ bias, float* __restrict__ out, int nRows)
{
  __shared__ __align__(16) unsigned int ldsA[64 * 64];   // 64 rows x 64 bf16 pairs
  __shared__ __align__(16) unsigned int ldsW[128 * 64];  // W^T tile: [n][kp]

  const int tid  = (int)threadIdx.x;
  const int lane = tid & 31;
  const int wave = tid >> 5;
  const int rowBase = (int)blockIdx.x * 64;
  const int mn = lane & 15;
  const int hi = lane >> 4;

  v8f acc[4];
  #pragma unroll
  for (int rt = 0; rt < 4; ++rt)
    #pragma unroll
    for (int j = 0; j < 8; ++j) acc[rt][j] = 0.0f;

  const int nPasses = (A1 != nullptr) ? 2 : 1;
  for (int pass = 0; pass < nPasses; ++pass) {
    const float* __restrict__ A = pass ? A1 : A0;
    const unsigned int* __restrict__ Wt = pass ? Wt1 : Wt0;

    // Stage Wt tile (32 KB raw copy): async global->LDS (ASYNCcnt path).
#if defined(HAVE_ASYNC_COPY)
    {
      g4p g = (g4p)(unsigned long long)(uintptr_t)Wt;
      l4p l = (l4p)(unsigned int)(uintptr_t)&ldsW[0];
      #pragma unroll
      for (int it = 0; it < 8; ++it) {
        int idx = it * 256 + tid;   // int4 (16B) granularity
        __builtin_amdgcn_global_load_async_to_lds_b128(g + idx, l + idx, 0, 0);
      }
    }
#else
    for (int i = tid; i < (128 * 64) / 4; i += 256) {
      ((uint4*)ldsW)[i] = ((const uint4*)Wt)[i];
    }
#endif

    // Stage A tile (64x128 fp32 -> packed bf16), coalesced float2 reads.
    for (int i = tid; i < 64 * 64; i += 256) {
      int r   = i >> 6;
      int kp  = i & 63;
      int row = rowBase + r;
      float x0 = 0.0f, x1 = 0.0f;
      if (row < nRows) {
        const float2 v = *(const float2*)(A + (size_t)row * F + kp * 2);
        x0 = v.x; x1 = v.y;
      }
      ldsA[i] = pack_bf16(x0, x1);
    }

#if defined(HAVE_ASYNC_COPY)
    wait_asynccnt0();
#endif
    __syncthreads();

    // B fragments (32x16 bf16 per K-chunk): v<4 -> K=2v+h (+8*hi), v>=4 -> 16+2(v-4)+h (+8*hi).
    v16bf bfrag[4];
    #pragma unroll
    for (int kc = 0; kc < 4; ++kc) {
      #pragma unroll
      for (int j = 0; j < 8; ++j) {
        int k = kc * 32 + hi * 8 + ((j < 4) ? (2 * j) : (16 + 2 * (j - 4)));
        union { unsigned int u; __bf16 h[2]; } p;
        p.u = ldsW[(wave * 16 + mn) * 64 + (k >> 1)];
        bfrag[kc][2 * j]     = p.h[0];
        bfrag[kc][2 * j + 1] = p.h[1];
      }
    }

    // A fragments (16x32 bf16): v<4 -> K=2v+h (+8*hi), v>=4 -> 16+2(v-4)+h (+8*hi).
    #pragma unroll
    for (int rt = 0; rt < 4; ++rt) {
      #pragma unroll
      for (int kc = 0; kc < 4; ++kc) {
        v16bf a;
        #pragma unroll
        for (int j = 0; j < 8; ++j) {
          int kb = (j < 4) ? (2 * j) : (16 + 2 * (j - 4));
          int k  = kc * 32 + hi * 8 + kb;
          union { unsigned int u; __bf16 h[2]; } p;
          p.u = ldsA[(rt * 16 + mn) * 64 + (k >> 1)];
          a[2 * j]     = p.h[0];
          a[2 * j + 1] = p.h[1];
        }
        acc[rt] = __builtin_amdgcn_wmma_f32_16x16x32_bf16(
            false, a, false, bfrag[kc], (short)0, acc[rt], false, false);
      }
    }
    __syncthreads();  // protect LDS before next pass restage
  }

  // Epilogue: bias + relu + store. D layout: VGPR r -> row r + 8*hi, col = lane&15.
  const int col = wave * 16 + mn;
  const float bv = bias[col];
  #pragma unroll
  for (int rt = 0; rt < 4; ++rt) {
    #pragma unroll
    for (int r = 0; r < 8; ++r) {
      int row = rowBase + rt * 16 + hi * 8 + r;
      if (row < nRows) {
        out[(size_t)row * F + col] = fmaxf(acc[rt][r] + bv, 0.0f);
      }
    }
  }
}

// agg[dst[e]][:] = max(agg[dst[e]][:], m[src[e]][:]) via uint-bit atomicMax
// (valid: m = relu(...) >= 0, agg zero-initialized; matches segment_max with
//  isolated-node -> 0 semantics).
__global__ __launch_bounds__(256) void edge_scatter_max_kernel(
    const float* __restrict__ m, const int* __restrict__ src,
    const int* __restrict__ dst, unsigned int* __restrict__ agg, int nEdges)
{
  const int lane = (int)threadIdx.x & 31;
  int e = (int)blockIdx.x * 8 + ((int)threadIdx.x >> 5);
  const int stride = (int)gridDim.x * 8;
  for (; e < nEdges; e += stride) {
    const int s = src[e];
    const int d = dst[e];
    if (e + stride < nEdges) {
      __builtin_prefetch(m + (size_t)src[e + stride] * F, 0, 0);
    }
    const float4 v = ((const float4*)(m + (size_t)s * F))[lane];
    unsigned int* ad = agg + (size_t)d * F + lane * 4;
    atomicMax(ad + 0, __float_as_uint(v.x));
    atomicMax(ad + 1, __float_as_uint(v.y));
    atomicMax(ad + 2, __float_as_uint(v.z));
    atomicMax(ad + 3, __float_as_uint(v.w));
  }
}

__global__ void fill_zero_kernel(unsigned int* __restrict__ p, long n) {
  long i = (long)blockIdx.x * blockDim.x + threadIdx.x;
  const long stride = (long)gridDim.x * blockDim.x;
  for (; i < n; i += stride) p[i] = 0u;
}

// Per-column max over h[nRows x 128] into gmax[128] (uint-bit max, h >= 0).
__global__ __launch_bounds__(256) void col_max_kernel(
    const float* __restrict__ h, unsigned int* __restrict__ gmax, int nRows)
{
  const int col = (int)threadIdx.x & 127;
  int r = (int)blockIdx.x * 2 + ((int)threadIdx.x >> 7);
  const int stride = (int)gridDim.x * 2;
  float mx = 0.0f;
  for (; r < nRows; r += stride) mx = fmaxf(mx, h[(size_t)r * F + col]);
  atomicMax(&gmax[col], __float_as_uint(mx));
}

// hc = [graph_max, states_max, action_row]; out = relu(hc@Wfc2+b)@Wfc3 + b.
__global__ __launch_bounds__(128) void head_kernel(
    const float* __restrict__ h, const unsigned int* __restrict__ gmax,
    const int* __restrict__ states, const int* __restrict__ actions,
    const float* __restrict__ Wfc2, const float* __restrict__ bfc2,
    const float* __restrict__ Wfc3, const float* __restrict__ bfc3,
    float* __restrict__ out)
{
  __shared__ float hc[3 * F];
  __shared__ float y[F];
  const int t = (int)threadIdx.x;
  hc[t] = __uint_as_float(gmax[t]);
  float smax = -INFINITY;
  #pragma unroll
  for (int i = 0; i < 8; ++i) smax = fmaxf(smax, h[(size_t)states[i] * F + t]);
  hc[F + t] = smax;
  hc[2 * F + t] = h[(size_t)actions[0] * F + t];
  __syncthreads();
  float acc = bfc2[t];
  for (int k = 0; k < 3 * F; ++k) acc += hc[k] * Wfc2[(size_t)k * F + t];
  acc = fmaxf(acc, 0.0f);
  y[t] = acc * Wfc3[t];
  __syncthreads();
  for (int s = 64; s > 0; s >>= 1) {
    if (t < s) y[t] += y[t + s];
    __syncthreads();
  }
  if (t == 0) out[0] = y[0] + bfc3[0];
}

extern "C" void kernel_launch(void* const* d_in, const int* in_sizes, int n_in,
                              void* d_out, int out_size, void* d_ws, size_t ws_size,
                              hipStream_t stream)
{
  (void)in_sizes; (void)n_in; (void)out_size; (void)ws_size;
  const float* inputs  = (const float*)d_in[0];
  const int*   src     = (const int*)d_in[1];
  const int*   dst     = (const int*)d_in[2];
  const int*   states  = (const int*)d_in[3];
  const int*   actions = (const int*)d_in[4];
  const float* Wp1     = (const float*)d_in[5];
  const float* bp1     = (const float*)d_in[6];
  const float* Wself1  = (const float*)d_in[7];
  const float* Wneigh1 = (const float*)d_in[8];
  const float* b1      = (const float*)d_in[9];
  const float* Wp2     = (const float*)d_in[10];
  const float* bp2     = (const float*)d_in[11];
  const float* Wself2  = (const float*)d_in[12];
  const float* Wneigh2 = (const float*)d_in[13];
  const float* b2      = (const float*)d_in[14];
  const float* Wfc2    = (const float*)d_in[15];
  const float* bfc2    = (const float*)d_in[16];
  const float* Wfc3    = (const float*)d_in[17];
  const float* bfc3    = (const float*)d_in[18];
  float* out = (float*)d_out;

  const size_t nodeF = (size_t)N_NODES * F;
  const size_t wElems = 128 * 64;       // packed uints per weight matrix
  float* bufM  = (float*)d_ws;          // m (pool output); reused as h2
  float* bufAG = bufM + nodeF;          // agg (scatter-max target)
  float* bufH1 = bufAG + nodeF;         // h1
  unsigned int* gmax = (unsigned int*)(bufH1 + nodeF);
  unsigned int* wt   = gmax + 128;      // 6 pre-transposed bf16 weight tiles
  unsigned int* WtP1 = wt + 0 * wElems;
  unsigned int* WtS1 = wt + 1 * wElems;
  unsigned int* WtN1 = wt + 2 * wElems;
  unsigned int* WtP2 = wt + 3 * wElems;
  unsigned int* WtS2 = wt + 4 * wElems;
  unsigned int* WtN2 = wt + 5 * wElems;

  // ---- One-shot weight transpose/convert (tiny) ----
  prep_weight_kernel<<<32, 256, 0, stream>>>(Wp1,     WtP1);
  prep_weight_kernel<<<32, 256, 0, stream>>>(Wself1,  WtS1);
  prep_weight_kernel<<<32, 256, 0, stream>>>(Wneigh1, WtN1);
  prep_weight_kernel<<<32, 256, 0, stream>>>(Wp2,     WtP2);
  prep_weight_kernel<<<32, 256, 0, stream>>>(Wself2,  WtS2);
  prep_weight_kernel<<<32, 256, 0, stream>>>(Wneigh2, WtN2);

  const int gemmGrid = (N_NODES + 63) / 64;

  // ---- Layer 1 ----
  sage_gemm_kernel<<<gemmGrid, 256, 0, stream>>>(inputs, WtP1, nullptr, nullptr, bp1, bufM, N_NODES);
  fill_zero_kernel<<<1024, 256, 0, stream>>>((unsigned int*)bufAG, (long)nodeF);
  edge_scatter_max_kernel<<<4096, 256, 0, stream>>>(bufM, src, dst, (unsigned int*)bufAG, N_EDGES);
  sage_gemm_kernel<<<gemmGrid, 256, 0, stream>>>(inputs, WtS1, bufAG, WtN1, b1, bufH1, N_NODES);

  // ---- Layer 2 ----
  sage_gemm_kernel<<<gemmGrid, 256, 0, stream>>>(bufH1, WtP2, nullptr, nullptr, bp2, bufM, N_NODES);
  fill_zero_kernel<<<1024, 256, 0, stream>>>((unsigned int*)bufAG, (long)nodeF);
  edge_scatter_max_kernel<<<4096, 256, 0, stream>>>(bufM, src, dst, (unsigned int*)bufAG, N_EDGES);
  sage_gemm_kernel<<<gemmGrid, 256, 0, stream>>>(bufH1, WtS2, bufAG, WtN2, b2, bufM, N_NODES);

  // ---- Head ----
  fill_zero_kernel<<<1, 128, 0, stream>>>(gmax, (long)F);
  col_max_kernel<<<1024, 256, 0, stream>>>(bufM, gmax, N_NODES);
  head_kernel<<<1, 128, 0, stream>>>(bufM, gmax, states, actions, Wfc2, bfc2, Wfc3, bfc3, out);
}
